// MultiHeadedAttention_72507637891396
// MI455X (gfx1250) — matrix operand order repaired and checked by reading, not compiled, not run on hardware
//
#include <hip/hip_runtime.h>

// ---------------------------------------------------------------------------
// MI455X (gfx1250) multi-head attention, bf16 WMMA pipeline + TDM staging.
//   B=4, S=2048, D_MODEL=512, H=8, D_K=64
// Stages:
//   1) Q/V projections -> bf16 [B,H,S,64]; K projection -> bf16 [B,H,64,S]
//      (K stored pre-transposed via LDS transpose in the GEMM epilogue)
//   2) flash attention: K^T / V tiles staged by TENSOR_LOAD_TO_LDS (TDM),
//      scores + PV via v_wmma_f32_16x16x32_bf16, online softmax in fp32
//   3) output projection -> fp32 out
// ---------------------------------------------------------------------------

typedef __attribute__((ext_vector_type(16))) __bf16 v16bf;
typedef __attribute__((ext_vector_type(8)))  __bf16 v8bf;
typedef __attribute__((ext_vector_type(8)))  float  v8f;
typedef __attribute__((ext_vector_type(4)))  unsigned int v4u;
typedef __attribute__((ext_vector_type(8)))  int v8i;
typedef __attribute__((ext_vector_type(4)))  int v4i;

union Frag { v16bf v; v8bf h[2]; };

__device__ __forceinline__ v8f wmma_bf16(const v16bf& a, const v16bf& b, const v8f& c) {
  return __builtin_amdgcn_wmma_f32_16x16x32_bf16(false, a, false, b, (short)0, c, false, false);
}

constexpr int BATCH = 4;
constexpr int SEQ   = 2048;
constexpr int DM    = 512;
constexpr int HEADS = 8;
constexpr int DK    = 64;
constexpr int MROWS = BATCH * SEQ;  // 8192

// -------------------------- TDM support ------------------------------------
#if defined(__HIP_DEVICE_COMPILE__) && defined(__has_builtin)
#  if __has_builtin(__builtin_amdgcn_tensor_load_to_lds)
#    define HAVE_TDM 1
#  endif
#endif
#ifndef HAVE_TDM
#  define HAVE_TDM 0
#endif

#if HAVE_TDM
// 2-D tile load: 64x64 bf16 tile, rows of 64 elems (128B) padded by 16B in
// LDS (pad_interval=32 DWORDs -> code 4, pad_amount=4 DWORDs -> code 3),
// giving the 144B row pitch the WMMA fragment readers use.
__device__ __forceinline__ void tdm_load_64x64(unsigned lds_off, const void* gptr,
                                               unsigned tdim0, unsigned tdim1,
                                               unsigned long long stride0) {
  const unsigned long long ga = (unsigned long long)(uintptr_t)gptr;
  v4u g0;
  g0[0] = 1u;                                                  // count=1
  g0[1] = lds_off;                                             // lds_addr
  g0[2] = (unsigned)(ga & 0xffffffffull);                      // global_addr lo
  g0[3] = ((unsigned)(ga >> 32) & 0x01ffffffu) | (2u << 30);   // addr hi | type=2
  v8i g1;
  g1[0] = (int)((1u << 16) | (1u << 20) | (4u << 22) | (3u << 25)); // 2B elems, pad on
  g1[1] = (int)((tdim0 & 0xffffu) << 16);                      // tensor_dim0 lo
  g1[2] = (int)(((tdim0 >> 16) & 0xffffu) | ((tdim1 & 0xffffu) << 16));
  g1[3] = (int)(((tdim1 >> 16) & 0xffffu) | (64u << 16));      // tile_dim0 = 64
  g1[4] = (int)64;                                             // tile_dim1 = 64
  g1[5] = (int)(unsigned)(stride0 & 0xffffffffull);            // tensor_dim0_stride
  g1[6] = (int)((unsigned)((stride0 >> 32) & 0xffffull));      // stride hi, dim1_stride=0
  g1[7] = 0;
  const v4i z = {0, 0, 0, 0};
#if __clang_major__ >= 23
  const v8i z8 = {0, 0, 0, 0, 0, 0, 0, 0};
  __builtin_amdgcn_tensor_load_to_lds(g0, g1, z, z, z8, 0);
#else
  __builtin_amdgcn_tensor_load_to_lds(g0, g1, z, z, 0);
#endif
}
#endif

// ---------------------------------------------------------------------------
// Generic K=512 GEMM:  Out[m,n] = sum_k A[m,k] * W[n,k] + bias[n]
//   A_F32    : A operand fp32 (convert to bf16 while staging) else bf16
//   OUT_MODE : 0 = fp32 [M,512]
//              1 = bf16 head layout [B,H,S,64]
//              2 = bf16 transposed head layout [B,H,64,S]  (for K)
// Tile: 128(M) x 64(N) x 32(K); 256 threads = 8 waves, wave w -> rows 16w..
// ---------------------------------------------------------------------------
template <bool A_F32, int OUT_MODE>
__global__ __launch_bounds__(256) void gemm512(const void* __restrict__ Aptr,
                                               const float* __restrict__ W,
                                               const float* __restrict__ bias,
                                               void* __restrict__ Out) {
  constexpr int TM = 128, TN = 64, TK = 32;
  constexpr int PA = TK + 8;  // 80B row pitch (16B aligned)
  constexpr int PB = TN + 8;  // 144B row pitch (16B aligned)
  __shared__ __bf16 Ab[TM * PA];  // [m][k]
  __shared__ __bf16 Bt[TK * PB];  // [k][n] (W staged transposed)

  const int tid  = threadIdx.x;
  const int lane = tid & 31;
  const int wave = tid >> 5;
  const int m0   = blockIdx.x * TM;
  const int n0   = blockIdx.y * TN;

  const int arow = tid >> 1;        // A staging: row
  const int acol = (tid & 1) * 16;  // 16 k-elems each
  const int brow = tid >> 2;        // W staging: n row
  const int bseg = (tid & 3) * 8;   // 8 k-elems each

  v8f acc[4] = {};

  for (int kt = 0; kt < DM; kt += TK) {
    __syncthreads();
    if constexpr (A_F32) {
      const float* ap = (const float*)Aptr + (size_t)(m0 + arow) * DM + kt + acol;
      v8bf lo, hi;
#pragma unroll
      for (int i = 0; i < 8; ++i) { lo[i] = (__bf16)ap[i]; hi[i] = (__bf16)ap[8 + i]; }
      *(v8bf*)&Ab[arow * PA + acol]     = lo;
      *(v8bf*)&Ab[arow * PA + acol + 8] = hi;
    } else {
      const __bf16* ap = (const __bf16*)Aptr + (size_t)(m0 + arow) * DM + kt + acol;
      *(v8bf*)&Ab[arow * PA + acol]     = *(const v8bf*)ap;
      *(v8bf*)&Ab[arow * PA + acol + 8] = *(const v8bf*)(ap + 8);
    }
    {
      const float* wp = W + (size_t)(n0 + brow) * DM + kt + bseg;
#pragma unroll
      for (int i = 0; i < 8; ++i) Bt[(bseg + i) * PB + brow] = (__bf16)wp[i];
    }
    __syncthreads();

    if (kt + TK < DM) {  // global_prefetch_b8 next K tiles
      if constexpr (A_F32)
        __builtin_prefetch((const float*)Aptr + (size_t)(m0 + arow) * DM + kt + TK + acol, 0, 0);
      else
        __builtin_prefetch((const __bf16*)Aptr + (size_t)(m0 + arow) * DM + kt + TK + acol, 0, 0);
      __builtin_prefetch(W + (size_t)(n0 + brow) * DM + kt + TK + bseg, 0, 0);
    }

    const int ar = wave * 16 + (lane & 15);
    const int kg = (lane >> 4) * 8;
    Frag a;
    a.h[0] = *(const v8bf*)&Ab[ar * PA + kg];
    a.h[1] = *(const v8bf*)&Ab[ar * PA + 16 + kg];
    Frag bm[4];
#pragma unroll
    for (int sub = 0; sub < 4; ++sub) {
#pragma unroll
      for (int i = 0; i < 8; ++i) {
        bm[sub].v[i]     = Bt[(kg + i) * PB + sub * 16 + (lane & 15)];
        bm[sub].v[8 + i] = Bt[(16 + kg + i) * PB + sub * 16 + (lane & 15)];
      }
    }
#pragma unroll
    for (int sub = 0; sub < 4; ++sub) acc[sub] = wmma_bf16(a.v, bm[sub].v, acc[sub]);
  }

  // ---- epilogue (C layout: VGPR r -> rows r / r+8 across lane halves) ----
  const int half = lane >> 4;
  const int col  = lane & 15;

  if constexpr (OUT_MODE == 2) {
    // K projection: transpose 128x64 tile through LDS, store [B,H,64,S] bf16
    constexpr int PT = TN + 8;
    __shared__ __bf16 Tp[TM * PT];
    __syncthreads();
#pragma unroll
    for (int sub = 0; sub < 4; ++sub) {
      const float bv = bias[n0 + sub * 16 + col];
#pragma unroll
      for (int r = 0; r < 8; ++r)
        Tp[(wave * 16 + r + half * 8) * PT + sub * 16 + col] = (__bf16)(acc[sub][r] + bv);
    }
    __syncthreads();
    const int n    = tid >> 2;          // 0..63 : output column of this tile
    const int rseg = (tid & 3) * 32;    // 32 rows each
    const int ng = n0 + n, h = ng >> 6, d = ng & (DK - 1);
    const int bi = m0 >> 11, s0 = (m0 & (SEQ - 1)) + rseg;
    __bf16* op = (__bf16*)Out + (((size_t)bi * HEADS + h) * DK + d) * SEQ + s0;
#pragma unroll
    for (int seg = 0; seg < 4; ++seg) {
      v8bf pk;
#pragma unroll
      for (int i = 0; i < 8; ++i) pk[i] = Tp[(rseg + seg * 8 + i) * PT + n];
      *(v8bf*)(op + seg * 8) = pk;
    }
  } else {
#pragma unroll
    for (int sub = 0; sub < 4; ++sub) {
      const int n = n0 + sub * 16 + col;
      const float bv = bias[n];
#pragma unroll
      for (int r = 0; r < 8; ++r) {
        const int m = m0 + wave * 16 + r + half * 8;
        const float y = acc[sub][r] + bv;
        if constexpr (OUT_MODE == 1) {
          const int bi = m >> 11, s = m & (SEQ - 1);
          const int h = n >> 6, d = n & (DK - 1);
          ((__bf16*)Out)[(((size_t)bi * HEADS + h) * SEQ + s) * DK + d] = (__bf16)y;
        } else {
          ((float*)Out)[(size_t)m * DM + n] = y;
        }
      }
    }
  }
}

// ---------------------------------------------------------------------------
// Flash attention. Block = 128 query rows x one (b,h); 8 waves, each owns 16
// full rows so the online softmax stays inside the wave (shfl butterflies over
// the 16-lane halves of the C layout). K^T/V tiles staged by the TDM.
// ---------------------------------------------------------------------------
__global__ __launch_bounds__(256) void attn_flash(const __bf16* __restrict__ Q,   // [B,H,S,64]
                                                  const __bf16* __restrict__ KT,  // [B,H,64,S]
                                                  const __bf16* __restrict__ V,   // [B,H,S,64]
                                                  const int* __restrict__ mask,   // [B,S,S]
                                                  __bf16* __restrict__ O) {       // [B,S,512]
  constexpr int PK = DK + 8;             // 144B pitch (matches TDM pad)
  __shared__ __bf16 Ks[DK * PK];         // [d][k] : K^T tile
  __shared__ __bf16 Vs[DK * PK];         // [k][d] : V tile
  __shared__ __bf16 Ps[8 * 16 * PK];     // per-wave P slab (C->A relayout)

  const int tid = threadIdx.x, lane = tid & 31, wave = tid >> 5;
  const int bh = blockIdx.y;             // 0..31
  const int b  = bh >> 3, h = bh & 7;
  const int q0 = blockIdx.x * 128;
  const int half = lane >> 4, col = lane & 15;
  const int kg = half * 8;

  // Q fragments for this wave's 16 rows, held in registers for all 32 k-tiles
  const __bf16* qbase = Q + ((size_t)bh * SEQ + q0 + wave * 16 + (lane & 15)) * DK;
  Frag qa[2];
#pragma unroll
  for (int ks = 0; ks < 2; ++ks) {
    qa[ks].h[0] = *(const v8bf*)(qbase + ks * 32 + kg);
    qa[ks].h[1] = *(const v8bf*)(qbase + ks * 32 + 16 + kg);
  }

  float mrow[8], lrow[8];
  v8f oacc[4] = {};
#pragma unroll
  for (int r = 0; r < 8; ++r) { mrow[r] = -3.0e38f; lrow[r] = 0.f; }

  const int* mbase = mask + (size_t)b * SEQ * SEQ;

#if !HAVE_TDM
  const int srow = tid >> 2;        // fallback staging: 64 rows x 16 elems
  const int sseg = (tid & 3) * 16;
#endif

  for (int kt = 0; kt < SEQ; kt += 64) {
    __syncthreads();
#if HAVE_TDM
    if (wave == 0) {
      // K^T tile: rows d (stride S), 64 contiguous k; V tile: rows k, 64 d
      tdm_load_64x64((unsigned)(uintptr_t)&Ks[0],
                     KT + (size_t)bh * DK * SEQ + kt, SEQ, DK, SEQ);
      tdm_load_64x64((unsigned)(uintptr_t)&Vs[0],
                     V + ((size_t)bh * SEQ + kt) * DK, DK, SEQ, DK);
      __builtin_amdgcn_s_wait_tensorcnt(0);
    }
#else
    {
      const __bf16* kp = KT + ((size_t)bh * DK + srow) * SEQ + kt + sseg;
      *(v8bf*)&Ks[srow * PK + sseg]     = *(const v8bf*)kp;
      *(v8bf*)&Ks[srow * PK + sseg + 8] = *(const v8bf*)(kp + 8);
      const __bf16* vp = V + ((size_t)bh * SEQ + kt + srow) * DK + sseg;
      *(v8bf*)&Vs[srow * PK + sseg]     = *(const v8bf*)vp;
      *(v8bf*)&Vs[srow * PK + sseg + 8] = *(const v8bf*)(vp + 8);
      if (kt + 64 < SEQ) {
        __builtin_prefetch(kp + 64, 0, 0);
        __builtin_prefetch(vp + 64 * DK, 0, 0);
      }
    }
#endif
    __syncthreads();

    // ---- scores S = Q K^T (16x64 per wave) ----
    v8f sacc[4] = {};
#pragma unroll
    for (int ks = 0; ks < 2; ++ks) {
      Frag bm[4];
#pragma unroll
      for (int sub = 0; sub < 4; ++sub) {
#pragma unroll
        for (int i = 0; i < 8; ++i) {
          bm[sub].v[i]     = Ks[(ks * 32 + kg + i) * PK + sub * 16 + col];
          bm[sub].v[8 + i] = Ks[(ks * 32 + 16 + kg + i) * PK + sub * 16 + col];
        }
      }
#pragma unroll
      for (int sub = 0; sub < 4; ++sub) sacc[sub] = wmma_bf16(qa[ks].v, bm[sub].v, sacc[sub]);
    }

    // ---- scale (1/sqrt(64)) + mask ----
#pragma unroll
    for (int sub = 0; sub < 4; ++sub)
#pragma unroll
      for (int r = 0; r < 8; ++r) {
        const int qrow = q0 + wave * 16 + r + half * 8;
        const int kcol = kt + sub * 16 + col;
        const int mv = mbase[(size_t)qrow * SEQ + kcol];
        const float s = sacc[sub][r] * 0.125f;
        sacc[sub][r] = (mv == 0) ? -1.0e9f : s;
      }

    // ---- online softmax per row (rows r / r+8 live in lane halves) ----
#pragma unroll
    for (int r = 0; r < 8; ++r) {
      float mx = fmaxf(fmaxf(sacc[0][r], sacc[1][r]), fmaxf(sacc[2][r], sacc[3][r]));
#pragma unroll
      for (int off = 1; off < 16; off <<= 1) mx = fmaxf(mx, __shfl_xor(mx, off, 16));
      const float mnew = fmaxf(mrow[r], mx);
      const float c = __expf(mrow[r] - mnew);
      mrow[r] = mnew;
      float rs = 0.f;
#pragma unroll
      for (int sub = 0; sub < 4; ++sub) {
        const float p = __expf(sacc[sub][r] - mnew);
        sacc[sub][r] = p;
        rs += p;
      }
#pragma unroll
      for (int off = 1; off < 16; off <<= 1) rs += __shfl_xor(rs, off, 16);
      lrow[r] = lrow[r] * c + rs;
#pragma unroll
      for (int sub = 0; sub < 4; ++sub) oacc[sub][r] *= c;
    }

    // ---- P: C layout -> A layout via wave-private LDS slab ----
    __bf16* Pw = &Ps[wave * 16 * PK];
#pragma unroll
    for (int sub = 0; sub < 4; ++sub)
#pragma unroll
      for (int r = 0; r < 8; ++r)
        Pw[(r + half * 8) * PK + sub * 16 + col] = (__bf16)sacc[sub][r];

    // ---- O += P V ----
#pragma unroll
    for (int ks = 0; ks < 2; ++ks) {
      Frag pa;
      pa.h[0] = *(const v8bf*)&Pw[(lane & 15) * PK + ks * 32 + kg];
      pa.h[1] = *(const v8bf*)&Pw[(lane & 15) * PK + ks * 32 + 16 + kg];
      Frag bm[4];
#pragma unroll
      for (int sub = 0; sub < 4; ++sub) {
#pragma unroll
        for (int i = 0; i < 8; ++i) {
          bm[sub].v[i]     = Vs[(ks * 32 + kg + i) * PK + sub * 16 + col];
          bm[sub].v[8 + i] = Vs[(ks * 32 + 16 + kg + i) * PK + sub * 16 + col];
        }
      }
#pragma unroll
      for (int sub = 0; sub < 4; ++sub) oacc[sub] = wmma_bf16(pa.v, bm[sub].v, oacc[sub]);
    }
  }

  // ---- epilogue: O / l -> bf16 [B, S, 512] (heads re-concatenated) ----
#pragma unroll
  for (int sub = 0; sub < 4; ++sub)
#pragma unroll
    for (int r = 0; r < 8; ++r) {
      const int s = q0 + wave * 16 + r + half * 8;
      const int n = h * DK + sub * 16 + col;
      O[((size_t)b * SEQ + s) * DM + n] = (__bf16)(oacc[sub][r] / lrow[r]);
    }
}

// ---------------------------------------------------------------------------
extern "C" void kernel_launch(void* const* d_in, const int* in_sizes, int n_in,
                              void* d_out, int out_size, void* d_ws, size_t ws_size,
                              hipStream_t stream) {
  const float* query = (const float*)d_in[0];
  const float* key   = (const float*)d_in[1];
  const float* value = (const float*)d_in[2];
  const int*   mask  = (const int*)d_in[3];
  const float* Wq = (const float*)d_in[4];  const float* bq = (const float*)d_in[5];
  const float* Wk = (const float*)d_in[6];  const float* bk = (const float*)d_in[7];
  const float* Wv = (const float*)d_in[8];  const float* bv = (const float*)d_in[9];
  const float* Wo = (const float*)d_in[10]; const float* bo = (const float*)d_in[11];
  float* out = (float*)d_out;

  const size_t headElems = (size_t)BATCH * HEADS * SEQ * DK;  // 4.19M elems
  char* ws = (char*)d_ws;
  __bf16* qh  = (__bf16*)(ws);                    // [B,H,S,64]
  __bf16* khT = (__bf16*)(ws + headElems * 2);    // [B,H,64,S]
  __bf16* vh  = (__bf16*)(ws + headElems * 4);    // [B,H,S,64]
  __bf16* ah  = (__bf16*)(ws + headElems * 6);    // [B,S,512]

  dim3 gp(MROWS / 128, DM / 64);  // 64 x 8
  gemm512<true, 1><<<gp, 256, 0, stream>>>(query, Wq, bq, qh);
  gemm512<true, 2><<<gp, 256, 0, stream>>>(key,   Wk, bk, khT);
  gemm512<true, 1><<<gp, 256, 0, stream>>>(value, Wv, bv, vh);

  dim3 ga(SEQ / 128, BATCH * HEADS);  // 16 x 32
  attn_flash<<<ga, 256, 0, stream>>>(qh, khT, vh, mask, ah);

  gemm512<false, 0><<<gp, 256, 0, stream>>>(ah, Wo, bo, out);
}
